// Transformer_3607772528860
// MI455X (gfx1250) — matrix operand order repaired and checked by reading, not compiled, not run on hardware
//
#include <hip/hip_runtime.h>
#include <hip/hip_bf16.h>

// ---------------------------------------------------------------------------
// MI455X (gfx1250) implementation.
// Precision: bf16 WMMA (v_wmma_f32_16x16x32_bf16) with f32 accumulation for
// all GEMMs; f32 VALU for LayerNorm / softmax.
// Post-attention chain (GEMM -> LN -> GEMM -> add -> LN) fused per row-block
// in LDS to avoid ~2 GB of HBM intermediates; only the 537 MB output is
// written to HBM.
// B-fragment sharing: each wave owns BOTH 16-row strips of the row-block, so
// every weight fragment fetched from L2 feeds two WMMAs (halves L2 traffic).
// ---------------------------------------------------------------------------

typedef __attribute__((ext_vector_type(16))) __bf16 v16bf;
typedef __attribute__((ext_vector_type(8)))  __bf16 v8bf;
typedef __attribute__((ext_vector_type(8)))  float  v8f;

#define EMBED  512
#define HEADS  8
#define HDIM   64
#define NTXT   512
#define NVID   512
#define NFRM   12
#define LN_EPS 1e-5f

#define SA_STR 520   // bf16 row stride (pad to dodge LDS bank conflicts)
#define SO_STR 520   // f32 row stride

static __device__ __forceinline__ v16bf frag_join(v8bf lo, v8bf hi) {
  return __builtin_shufflevector(lo, hi, 0,1,2,3,4,5,6,7,8,9,10,11,12,13,14,15);
}

// A fragment (16x32 bf16, M x K). base -> (row0, k0) of row-major [*, ld] array.
// ISA 7.12.2: lanes 0-15 row M=lane, K {0..7}U{16..23}; lanes 16-31 same M,
// K {8..15}U{24..31}.
static __device__ __forceinline__ v16bf load_a(const __bf16* base, int ld, int lane) {
  const __bf16* p = base + (lane & 15) * ld + ((lane >> 4) << 3);
  v8bf lo = *(const v8bf*)p;        // K = kb8 + 0..7
  v8bf hi = *(const v8bf*)(p + 16); // K = 16 + kb8 + 0..7
  return frag_join(lo, hi);
}

// B fragment (32x16, K x N) with B = W^T, i.e. B[k][n] = W[n][k], W row-major [*, ld].
// Lanes 0-15: K {0..7}U{16..23}, N=lane; lanes 16-31: K {8..15}U{24..31}, N=lane-16.
static __device__ __forceinline__ v16bf load_bT(const __bf16* w, int n0, int k0,
                                                int ld, int lane) {
  const __bf16* p = w + (size_t)(n0 + (lane & 15)) * ld + k0 + ((lane >> 4) << 3);
  v8bf lo = *(const v8bf*)p;
  v8bf hi = *(const v8bf*)(p + 16);
  return frag_join(lo, hi);
}

static __device__ __forceinline__ v8f wmma_bf16(v16bf a, v16bf b, v8f c) {
  return __builtin_amdgcn_wmma_f32_16x16x32_bf16(false, a, false, b, (short)0, c,
                                                 false, false);
}

static __device__ __forceinline__ v8f vzero8() {
  v8f z = {0.f,0.f,0.f,0.f,0.f,0.f,0.f,0.f};
  return z;
}

// ---------------------------------------------------------------------------
// f32 -> bf16 weight conversion (one-time per launch; weights stay L2-resident)
// ---------------------------------------------------------------------------
__global__ void cvt_bf16_kernel(const float* __restrict__ s, __bf16* __restrict__ d,
                                int n) {
  int i = (blockIdx.x * 256 + threadIdx.x) * 4;
  if (i + 3 < n) {
    float4 f = *(const float4*)(s + i);
    d[i]   = (__bf16)f.x;
    d[i+1] = (__bf16)f.y;
    d[i+2] = (__bf16)f.z;
    d[i+3] = (__bf16)f.w;
  }
}

// ---------------------------------------------------------------------------
// Row LayerNorm (E=512) -> bf16. 32 rows/block, 8 threads/row, shuffle reduce.
// ---------------------------------------------------------------------------
__global__ __launch_bounds__(256) void ln_rows_kernel(const float* __restrict__ x,
    const float* __restrict__ g, const float* __restrict__ b,
    __bf16* __restrict__ y) {
  int r  = blockIdx.x * 32 + (threadIdx.x >> 3);
  int c0 = (threadIdx.x & 7) * 64;
  const float* row = x + (size_t)r * EMBED + c0;
  float v[64];
  float s = 0.f, s2 = 0.f;
#pragma unroll
  for (int j = 0; j < 16; ++j) {
    float4 f = ((const float4*)row)[j];
    v[4*j+0] = f.x; v[4*j+1] = f.y; v[4*j+2] = f.z; v[4*j+3] = f.w;
  }
#pragma unroll
  for (int j = 0; j < 64; ++j) { s += v[j]; s2 += v[j] * v[j]; }
#pragma unroll
  for (int off = 1; off < 8; off <<= 1) {
    s  += __shfl_xor(s,  off);
    s2 += __shfl_xor(s2, off);
  }
  float mean = s * (1.f / EMBED);
  float rstd = rsqrtf(s2 * (1.f / EMBED) - mean * mean + LN_EPS);
  __bf16* yr = y + (size_t)r * EMBED + c0;
#pragma unroll
  for (int j = 0; j < 64; ++j)
    yr[j] = (__bf16)((v[j] - mean) * rstd * g[c0 + j] + b[c0 + j]);
}

// ---------------------------------------------------------------------------
// Projection GEMM: out[R,512] = A[R,512] @ W^T + bias   (bf16 in/out, f32 acc)
// 1 wave = one 32x16 strip (two M-tiles sharing each B fragment).
// 8 strips/block. Grid = (R/32)*32/8 = R/8 blocks.
// ---------------------------------------------------------------------------
__global__ __launch_bounds__(256) void proj_gemm_kernel(const __bf16* __restrict__ A,
    const __bf16* __restrict__ W, const float* __restrict__ bias,
    __bf16* __restrict__ out) {
  int lane = threadIdx.x & 31, wave = threadIdx.x >> 5;
  int strip = blockIdx.x * 8 + wave;
  int mb = strip >> 5;     // 32-row block
  int nt = strip & 31;     // col tile (N = 512 -> 32 tiles)
  v8f acc0 = vzero8(), acc1 = vzero8();
  const __bf16* a0base = A + (size_t)(mb * 32) * EMBED;
  const __bf16* a1base = a0base + (size_t)16 * EMBED;
  for (int k0 = 0; k0 < EMBED; k0 += 32) {
    v16bf a0 = load_a(a0base + k0, EMBED, lane);
    v16bf a1 = load_a(a1base + k0, EMBED, lane);
    v16bf b  = load_bT(W, nt * 16, k0, EMBED, lane);
    acc0 = wmma_bf16(a0, b, acc0);
    acc1 = wmma_bf16(a1, b, acc1);
  }
  int n  = nt * 16 + (lane & 15);
  int m0 = mb * 32 + ((lane >> 4) << 3);
  float bn = bias[n];
#pragma unroll
  for (int i = 0; i < 8; ++i) {
    out[(size_t)(m0 + i) * EMBED + n]      = (__bf16)(acc0[i] + bn);
    out[(size_t)(m0 + 16 + i) * EMBED + n] = (__bf16)(acc1[i] + bn);
  }
}

// ---------------------------------------------------------------------------
// Fused: per (video, 32-row text block):
//   attn (softmax over 12 frames, f32) -> LDS bf16
//   o = attn @ Wo^T + bo      (WMMA) -> LDS f32
//   a = LN2(o)                -> LDS f32 + bf16 copy
//   x = a + a @ Wl^T + bl     (WMMA)
//   out = LN3(x)              -> HBM f32
// Wave partition for GEMMs: wave w owns columns [w*64, w*64+64) and BOTH
// 16-row M-tiles; every B fragment feeds two WMMAs.
// ---------------------------------------------------------------------------
__global__ __launch_bounds__(256) void fused_attn_mlp_kernel(
    const __bf16* __restrict__ qg, const __bf16* __restrict__ kg,
    const __bf16* __restrict__ vg,
    const __bf16* __restrict__ wo, const float* __restrict__ bo,
    const __bf16* __restrict__ wl, const float* __restrict__ bl,
    const float* __restrict__ g2, const float* __restrict__ b2,
    const float* __restrict__ g3, const float* __restrict__ b3,
    float* __restrict__ out) {
  __shared__ alignas(16) unsigned char smem[32 * SA_STR * 2 + 32 * SO_STR * 4];
  __bf16* sA = (__bf16*)smem;                         // [32][SA_STR] bf16
  float*  sO = (float*)(smem + 32 * SA_STR * 2);      // [32][SO_STR] f32
  float*  sK = sO;                                    // [12][512] (attn phase only)
  float*  sV = sO + NFRM * EMBED;

  const int vid  = blockIdx.y;
  const int t0   = blockIdx.x * 32;
  const int tid  = threadIdx.x;
  const int lane = tid & 31;
  const int wave = tid >> 5;

  // ---- stage K,V (this video) into LDS as f32
  {
    const v8bf* kp = (const v8bf*)(kg + (size_t)vid * NFRM * EMBED);
    const v8bf* vp = (const v8bf*)(vg + (size_t)vid * NFRM * EMBED);
    for (int i = tid; i < NFRM * EMBED / 8; i += 256) {
      v8bf k8 = kp[i], v8 = vp[i];
#pragma unroll
      for (int j = 0; j < 8; ++j) {
        sK[i * 8 + j] = (float)k8[j];
        sV[i * 8 + j] = (float)v8[j];
      }
    }
  }
  __syncthreads();

  // ---- attention: one (t, head) per thread (32 rows x 8 heads = 256)
  {
    const int tl = tid & 31;
    const int h  = tid >> 5;
    float qd[64];
    const v8bf* qp = (const v8bf*)(qg + (size_t)(t0 + tl) * EMBED + h * HDIM);
#pragma unroll
    for (int c = 0; c < 8; ++c) {
      v8bf q8 = qp[c];
#pragma unroll
      for (int j = 0; j < 8; ++j) qd[c * 8 + j] = (float)q8[j];
    }
    float lg[NFRM];
    for (int f = 0; f < NFRM; ++f) {
      const float* kr = sK + f * EMBED + h * HDIM;
      float d = 0.f;
#pragma unroll
      for (int j = 0; j < 64; ++j) d += qd[j] * kr[j];
      lg[f] = d * 0.125f;  // 1/sqrt(64)
    }
    float m = lg[0];
#pragma unroll
    for (int f = 1; f < NFRM; ++f) m = fmaxf(m, lg[f]);
    float s = 0.f;
#pragma unroll
    for (int f = 0; f < NFRM; ++f) { lg[f] = __expf(lg[f] - m); s += lg[f]; }
    float rs = 1.f / s;
    float ac[64];
#pragma unroll
    for (int j = 0; j < 64; ++j) ac[j] = 0.f;
    for (int f = 0; f < NFRM; ++f) {
      float w = lg[f] * rs;
      const float* vr = sV + f * EMBED + h * HDIM;
#pragma unroll
      for (int j = 0; j < 64; ++j) ac[j] += w * vr[j];
    }
    __bf16* dst = sA + tl * SA_STR + h * HDIM;
#pragma unroll
    for (int j = 0; j < 64; ++j) dst[j] = (__bf16)ac[j];
  }
  __syncthreads();

  const int nbase = wave * 64;  // 4 n-tiles of 16 cols per wave

  // ---- GEMM1: o = attn @ Wo^T + bo  ->  sO (f32)
  {
    v8f acc0[4], acc1[4];
#pragma unroll
    for (int i = 0; i < 4; ++i) { acc0[i] = vzero8(); acc1[i] = vzero8(); }
    const __bf16* a0base = sA;                 // rows 0..15
    const __bf16* a1base = sA + 16 * SA_STR;   // rows 16..31
    for (int k0 = 0; k0 < EMBED; k0 += 32) {
      v16bf a0 = load_a(a0base + k0, SA_STR, lane);
      v16bf a1 = load_a(a1base + k0, SA_STR, lane);
#pragma unroll
      for (int nt = 0; nt < 4; ++nt) {
        v16bf b = load_bT(wo, nbase + nt * 16, k0, EMBED, lane);
        acc0[nt] = wmma_bf16(a0, b, acc0[nt]);
        acc1[nt] = wmma_bf16(a1, b, acc1[nt]);
      }
    }
    int m0 = (lane >> 4) << 3;
#pragma unroll
    for (int nt = 0; nt < 4; ++nt) {
      int n = nbase + nt * 16 + (lane & 15);
      float bn = bo[n];
#pragma unroll
      for (int i = 0; i < 8; ++i) {
        sO[(m0 + i) * SO_STR + n]      = acc0[nt][i] + bn;
        sO[(m0 + 16 + i) * SO_STR + n] = acc1[nt][i] + bn;
      }
    }
  }
  __syncthreads();

  // ---- LN2: attn_out f32 in-place in sO, bf16 copy in sA
  {
    int r  = tid >> 3;
    int c0 = (tid & 7) * 64;
    float x[64]; float s = 0.f, s2 = 0.f;
#pragma unroll
    for (int j = 0; j < 16; ++j) {
      float4 f = *(const float4*)(sO + r * SO_STR + c0 + 4 * j);
      x[4*j] = f.x; x[4*j+1] = f.y; x[4*j+2] = f.z; x[4*j+3] = f.w;
    }
#pragma unroll
    for (int j = 0; j < 64; ++j) { s += x[j]; s2 += x[j] * x[j]; }
#pragma unroll
    for (int off = 1; off < 8; off <<= 1) {
      s += __shfl_xor(s, off); s2 += __shfl_xor(s2, off);
    }
    float mean = s * (1.f / EMBED);
    float rstd = rsqrtf(s2 * (1.f / EMBED) - mean * mean + LN_EPS);
#pragma unroll
    for (int j = 0; j < 64; ++j) {
      float a = (x[j] - mean) * rstd * g2[c0 + j] + b2[c0 + j];
      sO[r * SO_STR + c0 + j] = a;
      sA[r * SA_STR + c0 + j] = (__bf16)a;
    }
  }
  __syncthreads();

  // ---- GEMM2: x = attn_out + attn_out @ Wl^T + bl  ->  sO (in place)
  {
    v8f acc0[4], acc1[4];
#pragma unroll
    for (int i = 0; i < 4; ++i) { acc0[i] = vzero8(); acc1[i] = vzero8(); }
    const __bf16* a0base = sA;
    const __bf16* a1base = sA + 16 * SA_STR;
    for (int k0 = 0; k0 < EMBED; k0 += 32) {
      v16bf a0 = load_a(a0base + k0, SA_STR, lane);
      v16bf a1 = load_a(a1base + k0, SA_STR, lane);
#pragma unroll
      for (int nt = 0; nt < 4; ++nt) {
        v16bf b = load_bT(wl, nbase + nt * 16, k0, EMBED, lane);
        acc0[nt] = wmma_bf16(a0, b, acc0[nt]);
        acc1[nt] = wmma_bf16(a1, b, acc1[nt]);
      }
    }
    int m0 = (lane >> 4) << 3;
#pragma unroll
    for (int nt = 0; nt < 4; ++nt) {
      int n = nbase + nt * 16 + (lane & 15);
      float bn = bl[n];
#pragma unroll
      for (int i = 0; i < 8; ++i) {
        int i0 = (m0 + i) * SO_STR + n;
        int i1 = (m0 + 16 + i) * SO_STR + n;
        sO[i0] = sO[i0] + acc0[nt][i] + bn;   // element owned by this lane only
        sO[i1] = sO[i1] + acc1[nt][i] + bn;
      }
    }
  }
  __syncthreads();

  // ---- LN3 + store to HBM
  {
    int r  = tid >> 3;
    int c0 = (tid & 7) * 64;
    float x[64]; float s = 0.f, s2 = 0.f;
#pragma unroll
    for (int j = 0; j < 16; ++j) {
      float4 f = *(const float4*)(sO + r * SO_STR + c0 + 4 * j);
      x[4*j] = f.x; x[4*j+1] = f.y; x[4*j+2] = f.z; x[4*j+3] = f.w;
    }
#pragma unroll
    for (int j = 0; j < 64; ++j) { s += x[j]; s2 += x[j] * x[j]; }
#pragma unroll
    for (int off = 1; off < 8; off <<= 1) {
      s += __shfl_xor(s, off); s2 += __shfl_xor(s2, off);
    }
    float mean = s * (1.f / EMBED);
    float rstd = rsqrtf(s2 * (1.f / EMBED) - mean * mean + LN_EPS);
    float* orow = out + ((size_t)vid * NTXT + (t0 + r)) * EMBED + c0;
#pragma unroll
    for (int j = 0; j < 64; ++j)
      orow[j] = (x[j] - mean) * rstd * g3[c0 + j] + b3[c0 + j];
  }
}

// ---------------------------------------------------------------------------
extern "C" void kernel_launch(void* const* d_in, const int* in_sizes, int n_in,
                              void* d_out, int out_size, void* d_ws, size_t ws_size,
                              hipStream_t stream) {
  const float* text  = (const float*)d_in[0];
  const float* video = (const float*)d_in[1];
  const float* Wq = (const float*)d_in[2];  const float* bq = (const float*)d_in[3];
  const float* Wk = (const float*)d_in[4];  const float* bk = (const float*)d_in[5];
  const float* Wv = (const float*)d_in[6];  const float* bv = (const float*)d_in[7];
  const float* Wo = (const float*)d_in[8];  const float* bo = (const float*)d_in[9];
  const float* Wl = (const float*)d_in[10]; const float* bl = (const float*)d_in[11];
  const float* g1 = (const float*)d_in[12]; const float* b1 = (const float*)d_in[13];
  const float* g2 = (const float*)d_in[14]; const float* b2 = (const float*)d_in[15];
  const float* g3 = (const float*)d_in[16]; const float* b3 = (const float*)d_in[17];
  float* out = (float*)d_out;

  // workspace layout (bf16): 5 weights + tn + q (512x512 each) + vn/k/v (6144x512)
  char* ws = (char*)d_ws;
  const size_t WB = (size_t)512 * 512 * 2;            // 512 KB
  const size_t VB = (size_t)6144 * 512 * 2;           // 6 MB
  __bf16* wqb = (__bf16*)(ws + 0 * WB);
  __bf16* wkb = (__bf16*)(ws + 1 * WB);
  __bf16* wvb = (__bf16*)(ws + 2 * WB);
  __bf16* wob = (__bf16*)(ws + 3 * WB);
  __bf16* wlb = (__bf16*)(ws + 4 * WB);
  __bf16* tn  = (__bf16*)(ws + 5 * WB);
  __bf16* qb  = (__bf16*)(ws + 6 * WB);
  __bf16* vn  = (__bf16*)(ws + 7 * WB);
  __bf16* kb  = (__bf16*)(ws + 7 * WB + VB);
  __bf16* vb  = (__bf16*)(ws + 7 * WB + 2 * VB);
  // total ws use: 7*0.5MB + 3*6MB = 21.5 MB

  const int NW = 512 * 512;  // elements per weight matrix
  cvt_bf16_kernel<<<256, 256, 0, stream>>>(Wq, wqb, NW);
  cvt_bf16_kernel<<<256, 256, 0, stream>>>(Wk, wkb, NW);
  cvt_bf16_kernel<<<256, 256, 0, stream>>>(Wv, wvb, NW);
  cvt_bf16_kernel<<<256, 256, 0, stream>>>(Wo, wob, NW);
  cvt_bf16_kernel<<<256, 256, 0, stream>>>(Wl, wlb, NW);

  ln_rows_kernel<<<512 / 32, 256, 0, stream>>>(text, g1, b1, tn);
  ln_rows_kernel<<<6144 / 32, 256, 0, stream>>>(video, g1, b1, vn);

  // grid = (R/32 strips)*(32 ntiles)/8 waves = R/8
  proj_gemm_kernel<<<512 / 8, 256, 0, stream>>>(tn, wqb, bq, qb);
  proj_gemm_kernel<<<6144 / 8, 256, 0, stream>>>(vn, wkb, bk, kb);
  proj_gemm_kernel<<<6144 / 8, 256, 0, stream>>>(vn, wvb, bv, vb);

  dim3 grid(512 / 32, 512);  // (text blocks, videos)
  fused_attn_mlp_kernel<<<grid, 256, 0, stream>>>(qb, kb, vb, wob, bo, wlb, bl,
                                                  g2, b2, g3, b3, out);
}
